// APPNPNet_2121713845071
// MI455X (gfx1250) — matrix operand order repaired and checked by reading, not compiled, not run on hardware
//
#include <hip/hip_runtime.h>
#include <hip/hip_bf16.h>
#include <stdint.h>

// ---------------- Problem constants (match reference) ----------------
#define N_NODES  100000
#define E_EDGES  1600000
#define IN_DIM   512
#define HIDDEN   512
#define OUT_DIM  48
#define K_STEPS  10
#define ALPHA    0.1f

#define BLK_ROWS 32          // node rows per workgroup (2 M-tiles)
#define LSF      516         // fp32 staging LDS row stride (16B aligned, bank-skewed)
#define LS       520         // bf16 tile LDS row stride (8-elem aligned, bank-skewed)

// ---------------- CDNA5 WMMA types ----------------
typedef __attribute__((ext_vector_type(16))) __bf16 v16bf;
typedef __attribute__((ext_vector_type(8)))  float  v8f;

struct B32x2 { uint4 lo; uint4 hi; };   // 32 bytes == v16bf

__device__ __forceinline__ unsigned short f2bf(float f) {
    unsigned int u = __float_as_uint(f);
    u += 0x7fffu + ((u >> 16) & 1u);     // round-to-nearest-even
    return (unsigned short)(u >> 16);
}

// A-matrix 16x32 bf16 fragment from a bf16 LDS tile (ISA 7.12.2 A-layout):
// lanes 0-15 : row M=lane,    K {k0+0..7,  k0+16..23}
// lanes 16-31: row M=lane-16, K {k0+8..15, k0+24..31}
__device__ __forceinline__ v16bf load_afrag(const unsigned short* base,
                                            int lu, int half, int k0) {
    const unsigned short* p = base + lu * LS + k0 + half * 8;
    B32x2 r;
    r.lo = *(const uint4*)(p);        // 8 bf16
    r.hi = *(const uint4*)(p + 16);   // 8 bf16, +16 elements
    return __builtin_bit_cast(v16bf, r);
}

// B-matrix 32x16 bf16 fragment from K-major (transposed) weights W[n][k]:
// lanes 0-15: col N=n0+lane, K {k0+0..7, k0+16..23} ; lanes 16-31: col, K {k0+8..15, k0+24..31}
__device__ __forceinline__ v16bf load_bfrag(const unsigned short* __restrict__ w,
                                            int lu, int half, int n0, int k0) {
    const unsigned short* p = w + (size_t)(n0 + lu) * HIDDEN + k0 + half * 8;
    B32x2 r;
    r.lo = *(const uint4*)(p);
    r.hi = *(const uint4*)(p + 16);
    return __builtin_bit_cast(v16bf, r);
}

// ---------------- Kernel 0: one-time weight transpose + bf16 convert ----------------
__global__ void convert_weights(const float* __restrict__ W1,
                                const float* __restrict__ W2,
                                unsigned short* __restrict__ W1T,
                                unsigned short* __restrict__ W2T) {
    int t = blockIdx.x * 256 + threadIdx.x;
    if (t >= HIDDEN * HIDDEN) return;
    int n = t >> 9;          // 0..511
    int k = t & 511;
    W1T[t] = f2bf(W1[(size_t)k * HIDDEN + n]);
    if (n < OUT_DIM)
        W2T[(size_t)n * HIDDEN + k] = f2bf(W2[(size_t)k * OUT_DIM + n]);
}

// ---------------- Kernel 1: fused MLP  h0 = relu(x@W1+b1)@W2 + b2 ----------------
// 32 node-rows per workgroup, 8 waves.
// Stage x (fp32) via async-to-LDS, convert once to a bf16 tile, then:
// Phase A: each wave owns 4 N-tiles x 2 M-tiles of h (B-fragment reuse = 2).
// Phase B: 6 waves own the 2x3 tiles of h0 (OUT_DIM = 48 = 3*16).
__global__ __launch_bounds__(256)
void mlp_fused(const float* __restrict__ x,
               const float* __restrict__ b1,
               const float* __restrict__ b2,
               const unsigned short* __restrict__ W1T,
               const unsigned short* __restrict__ W2T,
               float* __restrict__ h0,
               float* __restrict__ z) {
    // pool: fp32 staging tile (66KB) first, later reused as the bf16 h tile (33KB)
    __shared__ __align__(16) char pool[BLK_ROWS * LSF * 4];
    __shared__ unsigned short lxh[BLK_ROWS * LS];       // bf16 x tile
    float*          lxf = (float*)pool;                 // staging view (dead after cvt)
    unsigned short* lh  = (unsigned short*)pool;        // h tile view (phase A output)

    const int row0 = blockIdx.x * BLK_ROWS;
    const int tid  = threadIdx.x;

    // --- stage x[row0..row0+31][0..511] via GLOBAL_LOAD_ASYNC_TO_LDS_B128 ---
    // GVS mode: SGPR 64-bit base + per-lane 32-bit byte offset; dest = LDS addr VGPR.
    {
        const unsigned lbase = (unsigned)(uintptr_t)lxf;
        const unsigned long long gbase =
            (unsigned long long)(uintptr_t)(x + (size_t)row0 * IN_DIM);
        for (int i = tid; i < BLK_ROWS * (IN_DIM / 4); i += 256) {
            int row = i >> 7;                 // 128 float4-chunks per row
            int c4  = i & 127;
            unsigned laddr = lbase + (unsigned)((row * LSF + c4 * 4) * 4);
            unsigned goff  = (unsigned)((row * IN_DIM + c4 * 4) * 4);
            asm volatile("global_load_async_to_lds_b128 %0, %1, %2"
                         :: "v"(laddr), "v"(goff), "s"(gbase)
                         : "memory");
        }
        asm volatile("s_wait_asynccnt 0x0" ::: "memory");
    }
    __syncthreads();

    // --- one-shot f32 -> bf16 conversion pass (v_cvt_pk_bf16_f32, 64 elems/thread) ---
    for (int i = tid; i < BLK_ROWS * (IN_DIM / 4); i += 256) {
        int row = i >> 7;
        int c4  = i & 127;
        float4 v = *(const float4*)&lxf[row * LSF + c4 * 4];
        __bf16 b0 = (__bf16)v.x, b1v = (__bf16)v.y, b2v = (__bf16)v.z, b3 = (__bf16)v.w;
        uint2 pk;
        pk.x = (unsigned)__builtin_bit_cast(unsigned short, b0) |
               ((unsigned)__builtin_bit_cast(unsigned short, b1v) << 16);
        pk.y = (unsigned)__builtin_bit_cast(unsigned short, b2v) |
               ((unsigned)__builtin_bit_cast(unsigned short, b3) << 16);
        *(uint2*)&lxh[row * LS + c4 * 4] = pk;
    }
    __syncthreads();   // after this, lxf (pool) is dead; pool becomes lh

    const int wave = tid >> 5;
    const int lane = tid & 31;
    const int half = lane >> 4;       // 0: lanes 0-15, 1: lanes 16-31
    const int lu   = lane & 15;
    const int mb   = half * 8;        // C/D row base within a 16-row tile

    // ---------- Phase A: h = relu(x@W1 + b1); this wave: columns nt0..nt0+63 ----------
    const int nt0 = wave * 64;
    v8f acc[8] = {};                  // [t] rows 0-15, [4+t] rows 16-31
    for (int k0 = 0; k0 < HIDDEN; k0 += 32) {
        if (k0 + 32 < HIDDEN)   // pull next K-slab of B toward cache (global_prefetch_b8)
            __builtin_prefetch(&W1T[(size_t)(nt0 + lu) * HIDDEN + k0 + 32], 0, 1);
        v16bf a0 = load_afrag(lxh,            lu, half, k0);   // M 0-15
        v16bf a1 = load_afrag(lxh + 16 * LS,  lu, half, k0);   // M 16-31
#pragma unroll
        for (int t = 0; t < 4; ++t) {
            v16bf b = load_bfrag(W1T, lu, half, nt0 + t * 16, k0);
            acc[t]     = __builtin_amdgcn_wmma_f32_16x16x32_bf16(
                             false, a0, false, b, (short)0, acc[t],     false, false);
            acc[4 + t] = __builtin_amdgcn_wmma_f32_16x16x32_bf16(
                             false, a1, false, b, (short)0, acc[4 + t], false, false);
        }
    }
    // epilogue: bias + relu, write bf16 h tile to LDS (pool) per C-layout
#pragma unroll
    for (int t = 0; t < 4; ++t) {
        const int n = nt0 + t * 16 + lu;
        const float bias = b1[n];
#pragma unroll
        for (int r = 0; r < 8; ++r) {
            float v0 = acc[t][r] + bias;
            float v1 = acc[4 + t][r] + bias;
            v0 = v0 > 0.0f ? v0 : 0.0f;
            v1 = v1 > 0.0f ? v1 : 0.0f;
            lh[(mb + r) * LS + n]      = f2bf(v0);
            lh[(16 + mb + r) * LS + n] = f2bf(v1);
        }
    }
    __syncthreads();

    // ---------- Phase B: h0 = h@W2 + b2 (2 M-tiles x 3 N-tiles, waves 0..5) ----------
    if (wave < 6) {
        const int mt = wave / 3;          // M-tile: rows mt*16..mt*16+15
        const int n0 = (wave % 3) * 16;   // N-tile
        const unsigned short* lhm = lh + mt * 16 * LS;
        v8f c = {};
        for (int k0 = 0; k0 < HIDDEN; k0 += 32) {
            v16bf a = load_afrag(lhm, lu, half, k0);
            v16bf b = load_bfrag(W2T, lu, half, n0, k0);
            c = __builtin_amdgcn_wmma_f32_16x16x32_bf16(
                    false, a, false, b, (short)0, c, false, false);
        }
        const int n = n0 + lu;
        const float bias = b2[n];
#pragma unroll
        for (int r = 0; r < 8; ++r) {
            float v = c[r] + bias;
            size_t idx = (size_t)(row0 + mt * 16 + mb + r) * OUT_DIM + n;
            h0[idx] = v;   // teleport term (ALPHA * h0)
            z[idx]  = v;   // z starts as h0
        }
    }
}

// ---------------- Kernel 2: edge scatter  agg[row] += val * z[col] ----------------
// 16 lanes per edge, 3 dims per lane (48 = 16*3). z/agg/edges are L2-resident (~77MB).
__global__ __launch_bounds__(256)
void edge_spmm(const int* __restrict__ er, const int* __restrict__ ec,
               const float* __restrict__ ev,
               const float* __restrict__ z, float* __restrict__ agg) {
    long long t = (long long)blockIdx.x * 256 + threadIdx.x;
    int e  = (int)(t >> 4);
    int d0 = (int)(t & 15);
    if (e >= E_EDGES) return;
    const int   r = er[e];
    const int   c = ec[e];
    const float v = ev[e];
    const float* zs = z   + (size_t)c * OUT_DIM;
    float*       as = agg + (size_t)r * OUT_DIM;
#pragma unroll
    for (int j = 0; j < 3; ++j) {
        int d = d0 + j * 16;
        // non-returning relaxed agent-scope add -> global_atomic_add_f32 (STOREcnt path)
        __hip_atomic_fetch_add(&as[d], v * zs[d],
                               __ATOMIC_RELAXED, __HIP_MEMORY_SCOPE_AGENT);
    }
}

// ------- Kernel 3: z = (1-ALPHA)*agg + ALPHA*h0, and re-zero agg for next step -------
__global__ __launch_bounds__(256)
void axpby_clear(float* __restrict__ agg, const float* __restrict__ h0,
                 float* __restrict__ z, int n) {
    int i = blockIdx.x * 256 + threadIdx.x;
    if (i < n) {
        float a = agg[i];
        agg[i] = 0.0f;                                   // ready for next edge pass
        z[i] = (1.0f - ALPHA) * a + ALPHA * h0[i];
    }
}

// ---------------- Host-side launch ----------------
extern "C" void kernel_launch(void* const* d_in, const int* in_sizes, int n_in,
                              void* d_out, int out_size, void* d_ws, size_t ws_size,
                              hipStream_t stream) {
    (void)in_sizes; (void)n_in; (void)out_size; (void)ws_size;

    const float* x  = (const float*)d_in[0];
    const int*   er = (const int*)  d_in[1];
    const int*   ec = (const int*)  d_in[2];
    const float* ev = (const float*)d_in[3];
    const float* W1 = (const float*)d_in[4];
    const float* b1 = (const float*)d_in[5];
    const float* W2 = (const float*)d_in[6];
    const float* b2 = (const float*)d_in[7];
    float* z = (float*)d_out;                       // live z lives in d_out

    // workspace layout (all offsets 256B aligned)
    char* ws = (char*)d_ws;
    const size_t zbytes = (size_t)N_NODES * OUT_DIM * sizeof(float);  // 19.2 MB
    float*          h0  = (float*)(ws);
    float*          agg = (float*)(ws + zbytes);
    unsigned short* W1T = (unsigned short*)(ws + 2 * zbytes);
    unsigned short* W2T = (unsigned short*)(ws + 2 * zbytes + (size_t)HIDDEN * HIDDEN * 2);

    // 1) weight transpose + bf16 convert (one-time; W1T/W2T stay hot in L2)
    convert_weights<<<(HIDDEN * HIDDEN + 255) / 256, 256, 0, stream>>>(W1, W2, W1T, W2T);

    // 2) fused MLP via bf16 WMMA: writes h0 and initial z
    mlp_fused<<<N_NODES / BLK_ROWS, 256, 0, stream>>>(x, b1, b2, W1T, W2T, h0, z);

    // 3) APPNP propagation: K steps of SpMM + interpolation (agg self-clears each step)
    hipMemsetAsync(agg, 0, zbytes, stream);
    const long long edge_threads = (long long)E_EDGES * 16;
    const unsigned  edge_blocks  = (unsigned)((edge_threads + 255) / 256);
    const int       zn           = N_NODES * OUT_DIM;
    for (int s = 0; s < K_STEPS; ++s) {
        edge_spmm<<<edge_blocks, 256, 0, stream>>>(er, ec, ev, z, agg);
        axpby_clear<<<(zn + 255) / 256, 256, 0, stream>>>(agg, h0, z, zn);
    }
}
